// Encoder_Flows_6150393168179
// MI455X (gfx1250) — matrix operand order repaired and checked by reading, not compiled, not run on hardware
//
#include <hip/hip_runtime.h>

typedef __attribute__((ext_vector_type(16))) _Float16 v16h;
typedef __attribute__((ext_vector_type(8)))  _Float16 v8h;
typedef __attribute__((ext_vector_type(8)))  float    v8f;

#define K_ROWS 1024
#define EPS_N  1e-12f

// ---------------- fp32 -> f16 weight conversion ----------------
__global__ void cvt_f32_f16(const float* __restrict__ in, _Float16* __restrict__ out, int n) {
    int i = blockIdx.x * blockDim.x + threadIdx.x;
    int stride = gridDim.x * blockDim.x;
    for (; i < n; i += stride) out[i] = (_Float16)in[i];
}

// ---------------- prefix means over first 1024 rows ----------------
__global__ void sage_prefix(const float* __restrict__ x, _Float16* __restrict__ aggh, int din) {
    int j = blockIdx.x * blockDim.x + threadIdx.x;
    if (j >= din) return;
    float run = 0.f;
    for (int i = 0; i < K_ROWS; ++i) {
        float denom = (i > 0) ? (float)i : 1.f;
        aggh[(size_t)i * din + j] = (_Float16)(run / denom);
        run += x[(size_t)i * din + j];
    }
}

__device__ __forceinline__ v8h cvt8(const float* __restrict__ p) {
    float4 f0 = *(const float4*)(p);
    float4 f1 = *(const float4*)(p + 4);
    v8h h;
    h[0] = (_Float16)f0.x; h[1] = (_Float16)f0.y; h[2] = (_Float16)f0.z; h[3] = (_Float16)f0.w;
    h[4] = (_Float16)f1.x; h[5] = (_Float16)f1.y; h[6] = (_Float16)f1.z; h[7] = (_Float16)f1.w;
    return h;
}

__device__ __forceinline__ v16h join16(v8h lo, v8h hi) {
    v16h r;
    #pragma unroll
    for (int i = 0; i < 8; ++i) { r[i] = lo[i]; r[8 + i] = hi[i]; }
    return r;
}

// ---------------- fused GEMM + bias + agg-term + row L2-norm (+relu) ----------------
// Grid: N/64 row-blocks. Block: 256 threads = 8 waves.
// WG computes 64 rows x dout. Wave w owns N-tiles (w*TPW..w*TPW+TPW)*16, with 4 M-subtiles.
// A (64x32 f32 chunk of x) is cooperatively converted to f16 into LDS, double-buffered.
template<int TPW>
__global__ __launch_bounds__(256) void sage_gemm(
    const float* __restrict__ x, const _Float16* __restrict__ aggh,
    const _Float16* __restrict__ wlh, const float* __restrict__ bl,
    const _Float16* __restrict__ wrh, float* __restrict__ out,
    int din, int dout, int do_relu)
{
    __shared__ _Float16 sA[2][64 * 32];   // 2 x 4KB double buffer
    __shared__ float snorm[64];

    const int tid   = threadIdx.x;
    const int lane  = tid & 31;
    const int wave  = tid >> 5;
    const size_t rowbase = (size_t)blockIdx.x * 64;
    const int M     = lane & 15;                  // A row in subtile / C column in tile
    const int khalf = (lane >> 4) * 8;            // 16-bit A-matrix K sub-offset
    const int ksel  = (lane >> 4) * 16;           // added then cancelled in B address; B uses khalf

    // staging coords: thread -> (row 0..63, 8-wide K segment)
    const int srow = tid >> 2;
    const int sseg = (tid & 3) * 8;
    const float* xstage = x + (rowbase + srow) * (size_t)din + sseg;

    v8f acc[4][TPW] = {};
    const int nsteps = din >> 5;

    // stage chunk 0
    *(v8h*)&sA[0][srow * 32 + sseg] = cvt8(xstage);
    __syncthreads();

    for (int s = 0; s < nsteps; ++s) {
        const int kb = s << 5;
        if (s + 1 < nsteps)
            *(v8h*)&sA[(s + 1) & 1][srow * 32 + sseg] = cvt8(xstage + kb + 32);
        if (s + 2 < nsteps)
            __builtin_prefetch(xstage + kb + 64, 0, 1);

        // B tiles (f16 weights, 8 halves at khalf and 8 at 16+khalf per lane, same k order as A)
        v16h b[TPW];
        #pragma unroll
        for (int t = 0; t < TPW; ++t) {
            int col = (wave * TPW + t) * 16 + M;
            const _Float16* wr = wrh + (size_t)col * din + kb + ksel;
            b[t] = join16(*(const v8h*)(wr - ksel + khalf), *(const v8h*)(wr - ksel + 16 + khalf));
        }

        const _Float16* abuf = sA[s & 1];
        #pragma unroll
        for (int m = 0; m < 4; ++m) {
            const _Float16* ar = abuf + (m * 16 + M) * 32;
            v16h a = join16(*(const v8h*)(ar + khalf), *(const v8h*)(ar + 16 + khalf));
            #pragma unroll
            for (int t = 0; t < TPW; ++t)
                acc[m][t] = __builtin_amdgcn_wmma_f32_16x16x32_f16(
                    false, a, false, b[t], (short)0, acc[m][t], false, false);
        }
        __syncthreads();   // stage(s+1) visible; compute(s) done before buf reuse
    }

    // ---- agg term: agg @ w_l^T, only first 1024 rows (block-uniform branch) ----
    if (blockIdx.x < (K_ROWS / 64)) {
        for (int kb = 0; kb < din; kb += 32) {
            v16h b[TPW];
            #pragma unroll
            for (int t = 0; t < TPW; ++t) {
                int col = (wave * TPW + t) * 16 + M;
                const _Float16* wl = wlh + (size_t)col * din + kb + ksel;
                b[t] = join16(*(const v8h*)(wl - ksel + khalf), *(const v8h*)(wl - ksel + 16 + khalf));
            }
            #pragma unroll
            for (int m = 0; m < 4; ++m) {
                const _Float16* ar = aggh + (rowbase + m * 16 + M) * (size_t)din + kb;
                v16h a = join16(*(const v8h*)(ar + khalf), *(const v8h*)(ar + 16 + khalf));
                #pragma unroll
                for (int t = 0; t < TPW; ++t)
                    acc[m][t] = __builtin_amdgcn_wmma_f32_16x16x32_f16(
                        false, a, false, b[t], (short)0, acc[m][t], false, false);
            }
        }
    }

    // ---- epilogue: bias, 64-row L2 norms, store ----
    if (tid < 64) snorm[tid] = 0.f;
    __syncthreads();

    const int rbase = (lane >> 4) * 8;   // C/D: VGPR v holds row rbase+v within subtile
    float bias[TPW];
    #pragma unroll
    for (int t = 0; t < TPW; ++t) bias[t] = bl[(wave * TPW + t) * 16 + M];

    float p[4][8];
    #pragma unroll
    for (int m = 0; m < 4; ++m)
        #pragma unroll
        for (int v = 0; v < 8; ++v) p[m][v] = 0.f;

    #pragma unroll
    for (int m = 0; m < 4; ++m)
        #pragma unroll
        for (int t = 0; t < TPW; ++t)
            #pragma unroll
            for (int v = 0; v < 8; ++v) {
                float val = acc[m][t][v] + bias[t];
                acc[m][t][v] = val;
                p[m][v] += val * val;
            }

    #pragma unroll
    for (int mask = 1; mask <= 8; mask <<= 1)
        #pragma unroll
        for (int m = 0; m < 4; ++m)
            #pragma unroll
            for (int v = 0; v < 8; ++v)
                p[m][v] += __shfl_xor(p[m][v], mask, 32);

    if ((lane & 15) == 0) {
        #pragma unroll
        for (int m = 0; m < 4; ++m)
            #pragma unroll
            for (int v = 0; v < 8; ++v)
                atomicAdd(&snorm[m * 16 + rbase + v], p[m][v]);
    }
    __syncthreads();

    #pragma unroll
    for (int m = 0; m < 4; ++m)
        #pragma unroll
        for (int t = 0; t < TPW; ++t) {
            int col = (wave * TPW + t) * 16 + M;
            #pragma unroll
            for (int v = 0; v < 8; ++v) {
                int row = m * 16 + rbase + v;
                float inv = 1.f / fmaxf(sqrtf(snorm[row]), EPS_N);
                float val = acc[m][t][v] * inv;
                if (do_relu) val = fmaxf(val, 0.f);
                out[(rowbase + row) * (size_t)dout + col] = val;
            }
        }
}

__global__ void write_tail(float* out) {
    if (threadIdx.x == 0 && blockIdx.x == 0) out[0] = 1.0f;
}

extern "C" void kernel_launch(void* const* d_in, const int* in_sizes, int n_in,
                              void* d_out, int out_size, void* d_ws, size_t ws_size,
                              hipStream_t stream)
{
    (void)in_sizes; (void)n_in; (void)ws_size;
    const float* flow  = (const float*)d_in[0];
    const float* wl[4] = {(const float*)d_in[1], (const float*)d_in[4],
                          (const float*)d_in[7], (const float*)d_in[10]};
    const float* bl[4] = {(const float*)d_in[2], (const float*)d_in[5],
                          (const float*)d_in[8], (const float*)d_in[11]};
    const float* wr[4] = {(const float*)d_in[3], (const float*)d_in[6],
                          (const float*)d_in[9], (const float*)d_in[12]};

    const int N = 32 * 1024;                    // 32768 rows
    const int din_[4]  = {1024, 128, 256, 128};
    const int dout_[4] = {128, 256, 128, 256};

    char* ws = (char*)d_ws;
    size_t off = 0;
    auto alloc = [&](size_t bytes) -> char* {
        char* p = ws + off;
        off = (off + bytes + 255) & ~(size_t)255;
        return p;
    };
    _Float16* aggh = (_Float16*)alloc((size_t)K_ROWS * 1024 * 2);   // 2 MB max
    float* x1 = (float*)alloc((size_t)N * 128 * 4);                 // x1, reused as x3
    float* x2 = (float*)alloc((size_t)N * 256 * 4);
    _Float16* wlh[4]; _Float16* wrh[4];
    for (int l = 0; l < 4; ++l) {
        size_t e = (size_t)dout_[l] * din_[l];
        wlh[l] = (_Float16*)alloc(e * 2);
        wrh[l] = (_Float16*)alloc(e * 2);
    }

    // convert weights fp32 -> f16 once
    for (int l = 0; l < 4; ++l) {
        int e = dout_[l] * din_[l];
        int blocks = (e + 255) / 256;
        cvt_f32_f16<<<blocks, 256, 0, stream>>>(wl[l], wlh[l], e);
        cvt_f32_f16<<<blocks, 256, 0, stream>>>(wr[l], wrh[l], e);
    }

    const float* xin = flow;
    float* bufs[4] = {x1, x2, x1, (float*)d_out};
    for (int l = 0; l < 4; ++l) {
        int din = din_[l], dout = dout_[l];
        sage_prefix<<<(din + 127) / 128, 128, 0, stream>>>(xin, aggh, din);
        int relu = (l == 3) ? 1 : 0;
        if (dout == 128)
            sage_gemm<1><<<N / 64, 256, 0, stream>>>(xin, aggh, wlh[l], bl[l], wrh[l],
                                                     bufs[l], din, dout, relu);
        else
            sage_gemm<2><<<N / 64, 256, 0, stream>>>(xin, aggh, wlh[l], bl[l], wrh[l],
                                                     bufs[l], din, dout, relu);
        xin = bufs[l];
    }
    write_tail<<<1, 32, 0, stream>>>((float*)d_out + (out_size - 1));
}
